// LoRALinear_89558658056294
// MI455X (gfx1250) — matrix operand order repaired and checked by reading, not compiled, not run on hardware
//
#include <hip/hip_runtime.h>

// ---------------------------------------------------------------------------
// LoRA linear:  y = x @ W^T + b + 2.0 * ((x @ A^T) @ B^T)
//             = x @ (W + 2.0 * B@A)^T + b          (dropout = identity)
//
// Pass 0: W_eff = W + 2*B@A -> bf16 hi/lo planes in d_ws (4 MiB).
// Pass 1: X -> bf16 hi/lo planes in d_ws (128 MiB). Memory bound,
//         removes all convert VALU from the GEMM hot loop.
// Pass 2: GEMM, bf16x3 split (xh*wh + xh*wl + xl*wh), f32 accumulation.
//         ALL tile staging uses GLOBAL_LOAD_ASYNC_TO_LDS_B128 (ASYNCcnt),
//         fragments via ds_load_b128, math via v_wmma_f32_16x16x32_bf16.
//
// d_ws requirement: 2*D*D*2 + 2*M*D*2 = 132 MiB.
// ---------------------------------------------------------------------------

#define SCALING 2.0f

typedef __attribute__((ext_vector_type(16))) __bf16 v16bf;
typedef __attribute__((ext_vector_type(8)))  float  v8f;

constexpr int D        = 1024;   // in = out features
constexpr int BM       = 128;    // block tile M
constexpr int BN       = 128;    // block tile N
constexpr int BK       = 32;     // K chunk (== WMMA K)
constexpr int LDT      = BK + 8; // padded LDS row stride in halfwords
constexpr int NTHREADS = 256;    // 8 waves (wave32)

struct alignas(16) F4 { float x, y, z, w; };
struct alignas(8)  U2 { unsigned x, y; };
struct alignas(16) U4 { unsigned x, y, z, w; };
struct alignas(16) U8 { U4 a, b; };

__device__ __forceinline__ unsigned bfbits(__bf16 h) {
    return (unsigned)__builtin_bit_cast(unsigned short, h);
}

// split 4 consecutive f32 into packed bf16 hi/lo dword pairs
__device__ __forceinline__ void split4(F4 v, U2& H, U2& L) {
    __bf16 h0 = (__bf16)v.x, h1 = (__bf16)v.y, h2 = (__bf16)v.z, h3 = (__bf16)v.w;
    __bf16 l0 = (__bf16)(v.x - (float)h0);
    __bf16 l1 = (__bf16)(v.y - (float)h1);
    __bf16 l2 = (__bf16)(v.z - (float)h2);
    __bf16 l3 = (__bf16)(v.w - (float)h3);
    H.x = bfbits(h0) | (bfbits(h1) << 16);
    H.y = bfbits(h2) | (bfbits(h3) << 16);
    L.x = bfbits(l0) | (bfbits(l1) << 16);
    L.y = bfbits(l2) | (bfbits(l3) << 16);
}

// ---------------------------------------------------------------------------
// Pass 0: W_eff[n,k] = W[n,k] + 2*sum_r B[n,r]*A[r,k], emitted as bf16 hi/lo.
// ---------------------------------------------------------------------------
__global__ void build_weff(const float* __restrict__ W, const float* __restrict__ A,
                           const float* __restrict__ Bm,
                           unsigned short* __restrict__ Wh,
                           unsigned short* __restrict__ Wl) {
    int idx = blockIdx.x * NTHREADS + threadIdx.x;   // 0 .. D*D-1
    int n = idx >> 10;
    int k = idx & (D - 1);
    float acc = W[idx];
#pragma unroll
    for (int r = 0; r < 8; ++r)
        acc += SCALING * Bm[n * 8 + r] * A[r * D + k];
    __bf16 h = (__bf16)acc;
    __bf16 l = (__bf16)(acc - (float)h);
    Wh[idx] = (unsigned short)bfbits(h);
    Wl[idx] = (unsigned short)bfbits(l);
}

// ---------------------------------------------------------------------------
// Pass 1: split X into bf16 hi/lo planes (one float4 per thread).
// ---------------------------------------------------------------------------
__global__ void split_x(const float* __restrict__ X,
                        unsigned short* __restrict__ Xh,
                        unsigned short* __restrict__ Xl) {
    size_t q = (size_t)blockIdx.x * NTHREADS + threadIdx.x;  // float4 index
    F4 v = *(const F4*)(X + q * 4);
    U2 H, L;
    split4(v, H, L);
    *reinterpret_cast<U2*>(Xh + q * 4) = H;
    *reinterpret_cast<U2*>(Xl + q * 4) = L;
}

// ---------------------------------------------------------------------------
// Pass 2: Y[M,D] = X * Weff^T + bias.
// Grid (D/BN, M/BM); 8 waves as 4(M) x 2(N); wave tile 32x64 = 2x4 accums.
// ---------------------------------------------------------------------------
__global__ __launch_bounds__(NTHREADS)
void lora_gemm(const unsigned short* __restrict__ Xhg,
               const unsigned short* __restrict__ Xlg,
               const unsigned short* __restrict__ Whg,
               const unsigned short* __restrict__ Wlg,
               const float* __restrict__ bias, float* __restrict__ Y) {
    __shared__ unsigned short sXh[2][BM * LDT];
    __shared__ unsigned short sXl[2][BM * LDT];
    __shared__ unsigned short sWh[2][BN * LDT];
    __shared__ unsigned short sWl[2][BN * LDT];

    const int    tid    = threadIdx.x;
    const int    lane   = tid & 31;
    const int    wave   = tid >> 5;
    const int    wm     = wave & 3;      // 4 waves along M (32 rows each)
    const int    wn     = wave >> 2;     // 2 waves along N (64 cols each)
    const int    laneN  = lane & 15;
    const int    ksel   = lane >> 4;
    const size_t blockM = (size_t)blockIdx.y * BM;
    const int    blockN = blockIdx.x * BN;

    // one async 16B global->LDS transfer (no VGPR data round-trip)
    auto dma = [&](const unsigned short* g, unsigned short* s) {
        unsigned           ldsoff = (unsigned)(unsigned long long)(uintptr_t)s;
        unsigned long long gaddr  = (unsigned long long)(uintptr_t)g;
        asm volatile("global_load_async_to_lds_b128 %0, %1, off"
                     :: "v"(ldsoff), "v"(gaddr)
                     : "memory");
    };

    // stage one K-chunk: 4 bf16 planes, 128 rows x 32 halfwords each,
    // 8 async b128 transfers per thread.
    const int jj = tid & 3;              // 16B chunk within the 64B row
    const int r0 = tid >> 2;             // 0..63
    auto stage = [&](int buf, int kc) {
        const int k0 = kc * BK;
#pragma unroll
        for (int i = 0; i < 2; ++i) {
            const int    row = r0 + i * 64;
            const size_t gx  = (blockM + row) * D + k0 + jj * 8;
            const size_t gw  = ((size_t)blockN + row) * D + k0 + jj * 8;
            const int    so  = row * LDT + jj * 8;
            dma(Xhg + gx, &sXh[buf][so]);
            dma(Xlg + gx, &sXl[buf][so]);
            dma(Whg + gw, &sWh[buf][so]);
            dma(Wlg + gw, &sWl[buf][so]);
        }
    };

    // A fragment (16x32): lanes 0-15 -> K 0..7 & 16..23; lanes 16-31 -> +8
    auto fragA = [&](const unsigned short* s, int row) -> v16bf {
        const unsigned short* p = s + row * LDT;
        U8 t;
        t.a = *(const U4*)(p + ksel * 8);
        t.b = *(const U4*)(p + ksel * 8 + 16);
        return __builtin_bit_cast(v16bf, t);
    };
    // B fragment (32x16): lanes 0-15 -> K 0..7 & 16..23; lanes 16-31 -> +8
    auto fragB = [&](const unsigned short* s, int row) -> v16bf {
        const unsigned short* p = s + row * LDT;
        U8 t;
        t.a = *(const U4*)(p + ksel * 8);
        t.b = *(const U4*)(p + ksel * 8 + 16);
        return __builtin_bit_cast(v16bf, t);
    };

    v8f acc[2][4];
#pragma unroll
    for (int mi = 0; mi < 2; ++mi)
#pragma unroll
        for (int ni = 0; ni < 4; ++ni)
            acc[mi][ni] = (v8f){0.f, 0.f, 0.f, 0.f, 0.f, 0.f, 0.f, 0.f};

    constexpr int NK = D / BK;   // 32 chunks
    stage(0, 0);
    asm volatile("s_wait_asynccnt 0" ::: "memory");
    __syncthreads();

    for (int kc = 0; kc < NK; ++kc) {
        const int cur = kc & 1;
        if (kc + 1 < NK) stage(cur ^ 1, kc + 1);   // async prefetch next chunk

        v16bf ah[2], al[2], bh[4], bl[4];
#pragma unroll
        for (int mi = 0; mi < 2; ++mi) {
            int row = wm * 32 + mi * 16 + laneN;
            ah[mi] = fragA(sXh[cur], row);
            al[mi] = fragA(sXl[cur], row);
        }
#pragma unroll
        for (int ni = 0; ni < 4; ++ni) {
            int row = wn * 64 + ni * 16 + laneN;
            bh[ni] = fragB(sWh[cur], row);
            bl[ni] = fragB(sWl[cur], row);
        }

#pragma unroll
        for (int mi = 0; mi < 2; ++mi)
#pragma unroll
            for (int ni = 0; ni < 4; ++ni) {
                acc[mi][ni] = __builtin_amdgcn_wmma_f32_16x16x32_bf16(
                    false, ah[mi], false, bh[ni], (short)0, acc[mi][ni], false, false);
                acc[mi][ni] = __builtin_amdgcn_wmma_f32_16x16x32_bf16(
                    false, ah[mi], false, bl[ni], (short)0, acc[mi][ni], false, false);
                acc[mi][ni] = __builtin_amdgcn_wmma_f32_16x16x32_bf16(
                    false, al[mi], false, bh[ni], (short)0, acc[mi][ni], false, false);
            }

        asm volatile("s_wait_asynccnt 0" ::: "memory");  // DMA for next landed
        __syncthreads();
    }

    // Epilogue: C/D layout -> lane = N (laneN), VGPR j = row j + 8*ksel
#pragma unroll
    for (int mi = 0; mi < 2; ++mi)
#pragma unroll
        for (int ni = 0; ni < 4; ++ni) {
            int    n  = blockN + wn * 64 + ni * 16 + laneN;
            float  bn = bias[n];
            size_t m0 = blockM + wm * 32 + mi * 16 + ksel * 8;
#pragma unroll
            for (int j = 0; j < 8; ++j)
                Y[(m0 + j) * D + n] = acc[mi][ni][j] + bn;
        }
}

// ---------------------------------------------------------------------------
extern "C" void kernel_launch(void* const* d_in, const int* in_sizes, int n_in,
                              void* d_out, int out_size, void* d_ws, size_t ws_size,
                              hipStream_t stream) {
    const float* x  = (const float*)d_in[0];   // [B,S,D] f32
    const float* W  = (const float*)d_in[1];   // [D,D]
    const float* b  = (const float*)d_in[2];   // [D]
    const float* A  = (const float*)d_in[3];   // [R,D]
    const float* Bm = (const float*)d_in[4];   // [D,R]
    float*       y  = (float*)d_out;

    const int M = in_sizes[0] / D;             // 32768

    // d_ws layout: Wh | Wl | Xh | Xl  (132 MiB total)
    unsigned short* wh = (unsigned short*)d_ws;
    unsigned short* wl = wh + (size_t)D * D;
    unsigned short* xh = wl + (size_t)D * D;
    unsigned short* xl = xh + (size_t)M * D;

    build_weff<<<(D * D) / NTHREADS, NTHREADS, 0, stream>>>(W, A, Bm, wh, wl);
    split_x<<<(int)(((size_t)M * D / 4) / NTHREADS), NTHREADS, 0, stream>>>(x, xh, xl);

    dim3 grid(D / BN, M / BM);                 // (8, 256)
    lora_gemm<<<grid, NTHREADS, 0, stream>>>(xh, xl, wh, wl, b, y);
}
